// MSDeformAttn_87222195847397
// MI455X (gfx1250) — matrix operand order repaired and checked by reading, not compiled, not run on hardware
//
#include <hip/hip_runtime.h>

// ---------------------------------------------------------------------------
// MSDeformAttn forward for MI455X (gfx1250, wave32, WMMA)
//   N=4, LQ=13294, D=256, NH=8, NL=4, NP=4, CH=32, LIN=13294
// Pipeline:
//   1) value  = input_flatten @ W_val  + b_val   (WMMA GEMM, M=53176,K=256,N=256)
//   2) off    = query         @ W_off  + b_off   (WMMA GEMM, N=256)
//   3) attnlg = query         @ W_attn + b_attn  (WMMA GEMM, N=128)
//   4) sampled = deformable bilinear gather + softmax(attnlg) weighting
//   5) out    = sampled       @ W_out  + b_out   (WMMA GEMM, N=256)
// ---------------------------------------------------------------------------

typedef __attribute__((ext_vector_type(16))) _Float16 v16h;
typedef __attribute__((ext_vector_type(8)))  _Float16 h8;
typedef __attribute__((ext_vector_type(8)))  float    v8f;
typedef __attribute__((ext_vector_type(4)))  float    vf4;

#define N_BATCH 4
#define LQ      13294
#define D_MODEL 256
#define NH      8
#define NL      4
#define NPTS    4
#define CH      32
#define LIN     13294
#define M_TOT   (N_BATCH * LQ)   // 53176

#define BM 128
#define BN 64
#define BK 32
#define KTOT  256
#define NSTEP (KTOT / BK)        // 8, fully unrolled
#define LDA 40   // halves per row (80 B, multiple of 16 B for b128 LDS loads)
#define LDB 40

// ---------------------------------------------------------------------------
// GEMM: C[M,Nc] = A[M,256] * B[256,Nc] + bias[Nc], fp32 I/O, f16 WMMA core.
// 256 threads = 8 waves; block tile 128x64; each wave owns a 16-row strip and
// accumulates 4 v8f tiles (64 cols) via v_wmma_f32_16x16x32_f16 (32 static).
// Software-pipelined: tile s+1 global loads issue before tile s WMMAs.
// Row guard is a clamp (branch-free, EXEC stays all-ones for WMMA).
// ---------------------------------------------------------------------------
__global__ __launch_bounds__(256)
void gemm_bias_wmma(const float* __restrict__ A, const float* __restrict__ B,
                    const float* __restrict__ bias, float* __restrict__ C,
                    int M, int Nc)
{
  __shared__ alignas(16) _Float16 As[BM * LDA];   // [row][k]
  __shared__ alignas(16) _Float16 Bs[BN * LDB];   // [col][k] (transposed)

  const int tid  = threadIdx.x;
  const int lane = tid & 31;
  const int wave = tid >> 5;
  const int bm0  = blockIdx.x * BM;
  const int bn0  = blockIdx.y * BN;
  const int hsel = lane >> 4;      // 0: lanes 0-15, 1: lanes 16-31
  const int l15  = lane & 15;

  // ---- per-thread staging coordinates (A: 4 x vf4, B: 2 x vf4) ----
  int aRow[4], aC4[4], aGRow[4];
#pragma unroll
  for (int i = 0; i < 4; ++i) {
    const int f = tid + i * 256;              // 1024 vf4 = 128 rows * 8
    aRow[i] = f >> 3;
    aC4[i]  = (f & 7) << 2;
    int gr  = bm0 + aRow[i];
    aGRow[i] = gr < M ? gr : (M - 1);         // clamp: branch-free OOB guard
  }
  int bKr[2], bC4[2];
#pragma unroll
  for (int i = 0; i < 2; ++i) {
    const int f = tid + i * 256;              // 512 vf4 = 32 rows * 16
    bKr[i] = f >> 4;
    bC4[i] = (f & 15) << 2;
  }

  vf4 aReg[4], bReg[2];
  v8f acc[4] = {};

  // ---- prologue: stage tile 0 ----
#pragma unroll
  for (int i = 0; i < 4; ++i)
    aReg[i] = *(const vf4*)(A + (size_t)aGRow[i] * KTOT + aC4[i]);
#pragma unroll
  for (int i = 0; i < 2; ++i)
    bReg[i] = *(const vf4*)(B + (size_t)bKr[i] * Nc + bn0 + bC4[i]);
#pragma unroll
  for (int i = 0; i < 4; ++i) {
    _Float16* dst = &As[aRow[i] * LDA + aC4[i]];
    dst[0] = (_Float16)aReg[i][0]; dst[1] = (_Float16)aReg[i][1];
    dst[2] = (_Float16)aReg[i][2]; dst[3] = (_Float16)aReg[i][3];
  }
#pragma unroll
  for (int i = 0; i < 2; ++i)
#pragma unroll
    for (int j = 0; j < 4; ++j)
      Bs[(bC4[i] + j) * LDB + bKr[i]] = (_Float16)bReg[i][j];
  __syncthreads();

  const int ar  = wave * 16 + l15;
  const int kbA = hsel * 16;                   // A: lane<16 -> K0..15
  const int kbB = hsel * 16;                   // B: lane<16 -> K0..15

#pragma unroll
  for (int s = 0; s < NSTEP; ++s) {
    // ---- issue next tile's global loads (latency overlaps WMMAs) ----
    if (s + 1 < NSTEP) {
      const int k0n = (s + 1) * BK;
#pragma unroll
      for (int i = 0; i < 4; ++i)
        aReg[i] = *(const vf4*)(A + (size_t)aGRow[i] * KTOT + k0n + aC4[i]);
#pragma unroll
      for (int i = 0; i < 2; ++i)
        bReg[i] = *(const vf4*)(B + (size_t)(k0n + bKr[i]) * Nc + bn0 + bC4[i]);
      if (s + 2 < NSTEP)  // global_prefetch_b8 for the tile after next
        __builtin_prefetch(A + (size_t)aGRow[0] * KTOT + (s + 2) * BK, 0, 1);
    }

    // ---- fragments + 4 WMMAs from the staged tile ----
    h8 alo = *(const h8*)(&As[ar * LDA + kbA]);
    h8 ahi = *(const h8*)(&As[ar * LDA + kbA + 8]);
    v16h afrag = __builtin_shufflevector(alo, ahi,
        0,1,2,3,4,5,6,7,8,9,10,11,12,13,14,15);
#pragma unroll
    for (int c = 0; c < 4; ++c) {
      const int col = c * 16 + l15;
      h8 blo = *(const h8*)(&Bs[col * LDB + kbB]);
      h8 bhi = *(const h8*)(&Bs[col * LDB + kbB + 8]);
      v16h bfrag = __builtin_shufflevector(blo, bhi,
          0,1,2,3,4,5,6,7,8,9,10,11,12,13,14,15);
      acc[c] = __builtin_amdgcn_wmma_f32_16x16x32_f16(
          false, afrag, false, bfrag, (short)0, acc[c], false, false);
    }

    // ---- publish next tile into LDS ----
    if (s + 1 < NSTEP) {
      __syncthreads();
#pragma unroll
      for (int i = 0; i < 4; ++i) {
        _Float16* dst = &As[aRow[i] * LDA + aC4[i]];
        dst[0] = (_Float16)aReg[i][0]; dst[1] = (_Float16)aReg[i][1];
        dst[2] = (_Float16)aReg[i][2]; dst[3] = (_Float16)aReg[i][3];
      }
#pragma unroll
      for (int i = 0; i < 2; ++i)
#pragma unroll
        for (int j = 0; j < 4; ++j)
          Bs[(bC4[i] + j) * LDB + bKr[i]] = (_Float16)bReg[i][j];
      __syncthreads();
    }
  }

  // ---- store: C/D layout — VGPR r holds M = r + 8*hsel, N = l15 ----
#pragma unroll
  for (int c = 0; c < 4; ++c) {
    const int col = bn0 + c * 16 + l15;
    const float bv = bias[col];
#pragma unroll
    for (int r = 0; r < 8; ++r) {
      const int row = bm0 + wave * 16 + r + hsel * 8;
      if (row < M) C[(size_t)row * Nc + col] = acc[c][r] + bv;
    }
  }
}

// ---------------------------------------------------------------------------
// Deformable sampling: one wave per (n, q, h); lane = channel (CH == 32).
// Per-wave softmax over 16 logits, then 4 levels x 4 points x 4 bilinear
// taps of coalesced 128B gathers from value[N, LIN, NH, CH].
// ---------------------------------------------------------------------------
__global__ __launch_bounds__(256)
void deform_sample(const float* __restrict__ value,   // [N, LIN, NH, CH]
                   const float* __restrict__ refp,    // [N, LQ, NL, 2]
                   const float* __restrict__ off,     // [N, LQ, NH*NL*NP*2]
                   const float* __restrict__ alg,     // [N, LQ, NH*NL*NP]
                   const int*   __restrict__ shapes,  // [NL, 2] (H, W)
                   const int*   __restrict__ lstart,  // [NL]
                   float* __restrict__ out)           // [N, LQ, D]
{
  const int  lane = threadIdx.x & 31;
  const long wid  = ((long)blockIdx.x * blockDim.x + threadIdx.x) >> 5;
  const long nq   = wid >> 3;                  // (n, q) flat
  if (nq >= (long)M_TOT) return;
  const int h = (int)(wid & 7);
  const int n = (int)(nq / LQ);

  // softmax over the 16 (level, point) logits of this head
  const float* aptr = alg + nq * (NH * NL * NPTS) + h * (NL * NPTS);
  float w16[16];
  float mx = -3.4e38f;
#pragma unroll
  for (int j = 0; j < 16; ++j) mx = fmaxf(mx, aptr[j]);
  float sum = 0.f;
#pragma unroll
  for (int j = 0; j < 16; ++j) { w16[j] = __expf(aptr[j] - mx); sum += w16[j]; }
  const float inv = 1.f / sum;

  const float* optr = off  + nq * (NH * NL * NPTS * 2) + h * (NL * NPTS * 2);
  const float* rptr = refp + nq * (NL * 2);

  float res = 0.f;
#pragma unroll
  for (int l = 0; l < NL; ++l) {
    const int   Hs = shapes[2 * l], Ws = shapes[2 * l + 1];
    const int   st = lstart[l];
    const float fx = (float)Ws, fy = (float)Hs;
    const float rx = rptr[2 * l], ry = rptr[2 * l + 1];
    const float* vbase = value + (((size_t)n * LIN + st) * NH + h) * CH + lane;
#pragma unroll
    for (int p = 0; p < NPTS; ++p) {
      const float ox = optr[(l * NPTS + p) * 2];
      const float oy = optr[(l * NPTS + p) * 2 + 1];
      // loc = ref + off/[W,H]; pixel = loc*size - 0.5 (align_corners=False)
      const float x = (rx + ox / fx) * fx - 0.5f;
      const float y = (ry + oy / fy) * fy - 0.5f;
      const float x0f = floorf(x), y0f = floorf(y);
      const float wx = x - x0f, wy = y - y0f;
      const int   x0 = (int)x0f, y0 = (int)y0f;
      float ts = 0.f;
#pragma unroll
      for (int t = 0; t < 4; ++t) {
        const int dx = t & 1, dy = t >> 1;
        const int xi = x0 + dx, yi = y0 + dy;
        float w = (dx ? wx : 1.f - wx) * (dy ? wy : 1.f - wy);
        if (xi < 0 || xi >= Ws || yi < 0 || yi >= Hs) w = 0.f;   // zero padding
        const int xc = min(max(xi, 0), Ws - 1);
        const int yc = min(max(yi, 0), Hs - 1);
        const float g = vbase[(size_t)(yc * Ws + xc) * (NH * CH)];
        ts += w * g;
      }
      res += w16[l * NPTS + p] * inv * ts;
    }
  }
  out[nq * D_MODEL + h * CH + lane] = res;
}

// ---------------------------------------------------------------------------
extern "C" void kernel_launch(void* const* d_in, const int* in_sizes, int n_in,
                              void* d_out, int out_size, void* d_ws, size_t ws_size,
                              hipStream_t stream)
{
  const float* query  = (const float*)d_in[0];
  const float* refp   = (const float*)d_in[1];
  const float* inpf   = (const float*)d_in[2];
  const int*   shapes = (const int*)  d_in[3];
  const int*   lstart = (const int*)  d_in[4];
  const float* W_off  = (const float*)d_in[5];
  const float* b_off  = (const float*)d_in[6];
  const float* W_attn = (const float*)d_in[7];
  const float* b_attn = (const float*)d_in[8];
  const float* W_val  = (const float*)d_in[9];
  const float* b_val  = (const float*)d_in[10];
  const float* W_out  = (const float*)d_in[11];
  const float* b_out  = (const float*)d_in[12];
  float* out = (float*)d_out;

  // workspace partition (floats): value | off | attn logits | sampled
  float* ws = (float*)d_ws;
  const size_t sz256 = (size_t)M_TOT * 256;
  const size_t sz128 = (size_t)M_TOT * 128;
  float* v_val  = ws;
  float* v_off  = v_val + sz256;
  float* v_attn = v_off + sz256;
  float* v_samp = v_attn + sz128;

  const dim3 blk(256);
  const dim3 g256((M_TOT + BM - 1) / BM, 256 / BN);   // (416, 4)
  const dim3 g128((M_TOT + BM - 1) / BM, 128 / BN);   // (416, 2)

  // 1-3: projections
  gemm_bias_wmma<<<g256, blk, 0, stream>>>(inpf,  W_val,  b_val,  v_val,  M_TOT, 256);
  gemm_bias_wmma<<<g256, blk, 0, stream>>>(query, W_off,  b_off,  v_off,  M_TOT, 256);
  gemm_bias_wmma<<<g128, blk, 0, stream>>>(query, W_attn, b_attn, v_attn, M_TOT, 128);

  // 4: deformable bilinear sampling (8 waves / block, 1 wave per (n,q,h))
  const long waves   = (long)M_TOT * NH;          // 425,408
  const int  sblocks = (int)((waves + 7) / 8);    // 53,176
  deform_sample<<<sblocks, 256, 0, stream>>>(v_val, refp, v_off, v_attn,
                                             shapes, lstart, v_samp);

  // 5: output projection
  gemm_bias_wmma<<<g256, blk, 0, stream>>>(v_samp, W_out, b_out, out, M_TOT, 256);
}
